// GeneralLayer_69664369541268
// MI455X (gfx1250) — matrix operand order repaired and checked by reading, not compiled, not run on hardware
//
#include <hip/hip_runtime.h>

// ---------------------------------------------------------------------------
// GeneralLayer fused forward for MI455X (gfx1250, wave32, WMMA).
//
// Inputs (setup_inputs order, all float32):
//   d_in[0] x      (32768, 256)
//   d_in[1] PL     (512, 256)
//   d_in[2] PR     (512, 256)
//   d_in[3] Wunits (512, 6)
//   d_in[4] WL     (256, 512)
// Output: x_next (32768, 256) float32.
//
// Math (boolean-gate collapse): outs = C0[s] + C1[s]*(a+b) + C2[s]*a*b,
// a = x@PLs^T, b = x@PRs^T, x_next = outs @ Lrows^T.  All three GEMMs use
// v_wmma_f32_16x16x32_f16 (operands in [0,1], f16 + f32 accumulate).
//
// Main kernel: 64 batch rows per WG, 8 wave32s, wave grid 2(Mpair) x 4(N).
//   - x tile staged once into LDS (f32, 64 KB) via async-to-LDS when available
//   - per K-half (256 s-units): phase1 -> outs f16 tile (32 KB LDS),
//     barrier, phase2 accumulates into persistent v8f acc[8]
//   - each B fragment feeds 2 WMMAs (2 M-tiles per wave) to halve L2 traffic
// Static LDS: 96 KB (probes gfx1250 >64KB WGP LDS).
// ---------------------------------------------------------------------------

#define BATCH 32768
#define B_IN  256
#define S_UNITS 512
#define OUT_BITS 256
#define INV_TAU (1.0f / 0.3f)

typedef __attribute__((ext_vector_type(16))) _Float16 v16h;
typedef __attribute__((ext_vector_type(8)))  float    v8f;
typedef __attribute__((ext_vector_type(4)))  int      v4i;

struct U32B { uint4 a, b; };
union  H16  { v16h v; unsigned short s[16]; };

#if __has_builtin(__builtin_amdgcn_global_load_async_to_lds_b128) && \
    __has_builtin(__builtin_amdgcn_s_wait_asynccnt)
#define USE_ASYNC_LDS 1
#else
#define USE_ASYNC_LDS 0
#endif

static __device__ __forceinline__ unsigned short f2h(float f) {
    _Float16 h = (_Float16)f;
    return __builtin_bit_cast(unsigned short, h);
}

// 16 contiguous f16 (B fragments: col = lane%16, K = kbase..kbase+15).
static __device__ __forceinline__ v16h load16h(const unsigned short* p) {
    U32B u;
    u.a = *(const uint4*)(p);
    u.b = *(const uint4*)(p + 8);
    return __builtin_bit_cast(v16h, u);
}

// A fragment from f16 row-major storage: elements 0..15 = K0..K0+15
// (contiguous K order, matching the B fragments).
static __device__ __forceinline__ v16h load_afrag(const unsigned short* p) {
    U32B u;
    u.a = *(const uint4*)(p);
    u.b = *(const uint4*)(p + 8);
    return __builtin_bit_cast(v16h, u);
}

static __device__ __forceinline__ v8f wmma16(v16h a, v16h b, v8f c) {
    // (neg_a, A, neg_b, B, c_mod, C, reuse_a, reuse_b)
    return __builtin_amdgcn_wmma_f32_16x16x32_f16(false, a, false, b,
                                                  (short)0, c, false, false);
}

// ---------------------------------------------------------------------------
// Prep kernel 1: per pair-unit s (512 blocks x 256 threads).
// PLs softmax (+argmax), repel-gated PRs softmax, boolean-gate collapse.
// ---------------------------------------------------------------------------
__global__ __launch_bounds__(256) void prep_pairs(
        const float* __restrict__ PL, const float* __restrict__ PR,
        const float* __restrict__ Wunits,
        unsigned short* __restrict__ Wa, unsigned short* __restrict__ Wb,
        float* __restrict__ C0, float* __restrict__ C1, float* __restrict__ C2) {
    const int s = blockIdx.x;
    const int t = threadIdx.x;
    __shared__ float red[256];
    __shared__ int   redi[256];

    // ---- PLs = softmax(PL/tau), with first-max argmax ----
    const float v = PL[s * B_IN + t] * INV_TAU;
    red[t] = v; redi[t] = t;
    __syncthreads();
    for (int off = 128; off > 0; off >>= 1) {
        if (t < off) {
            float o = red[t + off]; int oi = redi[t + off];
            if (o > red[t] || (o == red[t] && oi < redi[t])) { red[t] = o; redi[t] = oi; }
        }
        __syncthreads();
    }
    const float mx = red[0];
    const int amax = redi[0];
    __syncthreads();

    const float e = __expf(v - mx);
    red[t] = e;
    __syncthreads();
    for (int off = 128; off > 0; off >>= 1) {
        if (t < off) red[t] += red[t + off];
        __syncthreads();
    }
    const float pls = e / red[0];
    __syncthreads();
    Wa[s * B_IN + t] = f2h(pls);

    // ---- PRs = softmax(PR/tau + log(clip(1-PLs,eps,1)) + mask) ----
    const float gate = __logf(fminf(fmaxf(1.0f - pls, 1e-6f), 1.0f));
    float r = PR[s * B_IN + t] * INV_TAU + gate;   // ETA = 1.0
    const bool masked = (t == amax);
    if (masked) r = -__builtin_inff();

    red[t] = r;
    __syncthreads();
    for (int off = 128; off > 0; off >>= 1) {
        if (t < off) red[t] = fmaxf(red[t], red[t + off]);
        __syncthreads();
    }
    const float mx2 = red[0];
    __syncthreads();
    const float e2 = masked ? 0.0f : __expf(r - mx2);
    red[t] = e2;
    __syncthreads();
    for (int off = 128; off > 0; off >>= 1) {
        if (t < off) red[t] += red[t + off];
        __syncthreads();
    }
    Wb[s * B_IN + t] = f2h(e2 / red[0]);

    // ---- boolean-unit collapse: outs = C0 + C1*(a+b) + C2*a*b ----
    if (t == 0) {
        float w[6], p[6];
        float wmax = -__builtin_inff();
        for (int i = 0; i < 6; ++i) { w[i] = Wunits[s * 6 + i] * INV_TAU; wmax = fmaxf(wmax, w[i]); }
        float sum = 0.0f;
        for (int i = 0; i < 6; ++i) { p[i] = __expf(w[i] - wmax); sum += p[i]; }
        for (int i = 0; i < 6; ++i) p[i] /= sum;
        C0[s] = p[3] + p[4] + p[5];
        C1[s] = p[1] + p[2] - p[4] - p[5];
        C2[s] = p[0] - p[1] - 2.0f * p[2] - p[3] + p[4] + 2.0f * p[5];
    }
}

// ---------------------------------------------------------------------------
// Prep kernel 2: L_rows = softmax(WL/tau) per out row (256 blocks x 256 thr).
// ---------------------------------------------------------------------------
__global__ __launch_bounds__(256) void prep_wl(
        const float* __restrict__ WL, unsigned short* __restrict__ WLb) {
    const int o = blockIdx.x;
    const int t = threadIdx.x;
    __shared__ float red[256];

    const float v0 = WL[o * S_UNITS + t]       * INV_TAU;
    const float v1 = WL[o * S_UNITS + 256 + t] * INV_TAU;
    red[t] = fmaxf(v0, v1);
    __syncthreads();
    for (int off = 128; off > 0; off >>= 1) {
        if (t < off) red[t] = fmaxf(red[t], red[t + off]);
        __syncthreads();
    }
    const float mx = red[0];
    __syncthreads();
    const float e0 = __expf(v0 - mx), e1 = __expf(v1 - mx);
    red[t] = e0 + e1;
    __syncthreads();
    for (int off = 128; off > 0; off >>= 1) {
        if (t < off) red[t] += red[t + off];
        __syncthreads();
    }
    const float inv = 1.0f / red[0];
    WLb[o * S_UNITS + t]       = f2h(e0 * inv);
    WLb[o * S_UNITS + 256 + t] = f2h(e1 * inv);
}

// ---------------------------------------------------------------------------
// Fused main kernel.
// Wave grid: mgrp = wave&1 -> rows mgrp*32..+31 (2 M-tiles),
//            nq   = wave>>1 -> 16 s-cols (phase 1) / 64 out-cols (phase 2).
// ---------------------------------------------------------------------------
__global__ __launch_bounds__(256) void glayer_fused(
        const float* __restrict__ x,
        const unsigned short* __restrict__ Wa,
        const unsigned short* __restrict__ Wb,
        const unsigned short* __restrict__ WLb,
        const float* __restrict__ C0,
        const float* __restrict__ C1,
        const float* __restrict__ C2,
        float* __restrict__ out) {
    __shared__ float          lds_xf[64 * 256];    // 64 KB  x tile (f32)
    __shared__ unsigned short lds_outs[64 * 256];  // 32 KB  outs K-half (f16)

    const int tid    = threadIdx.x;
    const int wave   = tid >> 5;
    const int lane   = tid & 31;
    const int lanelo = lane & 15;
    const int lanehi = lane >> 4;
    const int mgrp   = wave & 1;     // 32-row group
    const int nq     = wave >> 1;    // 0..3
    const int rowbase = blockIdx.x * 64;

    // ---- stage x tile into LDS (async-to-LDS when the builtin exists) ----
    {
        const float* xbase = x + rowbase * B_IN;
        #pragma unroll 1
        for (int it = 0; it < 16; ++it) {
            const int i   = tid + it * 256;          // float4 index in tile
            const int row = i >> 6;
            const int c4  = (i & 63) * 4;
            const float* gp = xbase + row * B_IN + c4;
            float*       lp = &lds_xf[row * B_IN + c4];
#if USE_ASYNC_LDS
            __builtin_amdgcn_global_load_async_to_lds_b128(
                (__attribute__((address_space(1))) v4i*)gp,
                (__attribute__((address_space(3))) v4i*)lp, 0, 0);
#else
            *(float4*)lp = *(const float4*)gp;
#endif
        }
#if USE_ASYNC_LDS
        __builtin_amdgcn_s_wait_asynccnt(0);
#endif
        __syncthreads();
    }

    v8f acc[8] = {};   // phase-2 accumulators: [mt*4+nt], persistent over halves

    #pragma unroll 1
    for (int half = 0; half < 2; ++half) {           // 256 s-units per half
        // ============ Phase 1: a/b GEMMs + boolean combine ============
        #pragma unroll 1
        for (int chunk = 0; chunk < 4; ++chunk) {    // 64 s-cols per chunk
            const int sl = chunk * 64 + nq * 16 + lanelo;   // s within half
            const int sg = half * 256 + sl;                 // global s

            if (chunk < 3) {   // prefetch next chunk's weight rows
                const int ps = sg + 64;
                __builtin_prefetch(Wa + ps * B_IN, 0, 1);
                __builtin_prefetch(Wb + ps * B_IN, 0, 1);
            }

            v8f accA0 = {}, accA1 = {}, accB0 = {}, accB1 = {};
            #pragma unroll
            for (int ks = 0; ks < 8; ++ks) {         // K = 256, step 32
                // A fragments (2 M-tiles) from LDS x, f32 -> f16
                const int k0 = ks * 32 + lanehi * 16;
                v16h afr[2];
                #pragma unroll
                for (int mt = 0; mt < 2; ++mt) {
                    const float* xp =
                        &lds_xf[(mgrp * 32 + mt * 16 + lanelo) * B_IN + k0];
                    float tf[16];
                    *(float4*)(tf)      = *(const float4*)(xp);
                    *(float4*)(tf + 4)  = *(const float4*)(xp + 4);
                    *(float4*)(tf + 8)  = *(const float4*)(xp + 8);
                    *(float4*)(tf + 12) = *(const float4*)(xp + 12);
                    H16 h;
                    #pragma unroll
                    for (int i = 0; i < 16; ++i) h.s[i] = f2h(tf[i]);
                    afr[mt] = h.v;
                }
                // B fragments: PLs / PRs rows; each feeds 2 WMMAs
                const int kb = ks * 32 + lanehi * 16;
                const v16h bA = load16h(Wa + sg * B_IN + kb);
                const v16h bB = load16h(Wb + sg * B_IN + kb);
                accA0 = wmma16(afr[0], bA, accA0);
                accA1 = wmma16(afr[1], bA, accA1);
                accB0 = wmma16(afr[0], bB, accB0);
                accB1 = wmma16(afr[1], bB, accB1);
            }

            // combine: outs = C0 + C1*(a+b) + C2*a*b -> f16 LDS tile
            const float c0 = C0[sg], c1 = C1[sg], c2 = C2[sg];
            #pragma unroll
            for (int mt = 0; mt < 2; ++mt) {
                const v8f a = mt ? accA1 : accA0;
                const v8f b = mt ? accB1 : accB0;
                #pragma unroll
                for (int r = 0; r < 8; ++r) {        // M = r + 8*(lane>=16)
                    const float o = c0 + c1 * (a[r] + b[r]) + c2 * (a[r] * b[r]);
                    const int m = mgrp * 32 + mt * 16 + r + lanehi * 8;
                    lds_outs[m * 256 + sl] = f2h(o);
                }
            }
        }
        __syncthreads();   // outs K-half complete

        // ============ Phase 2: accumulate outs @ L^T over this K-half ======
        #pragma unroll 1
        for (int ks = 0; ks < 8; ++ks) {             // K-half = 256, step 32
            const int k0 = ks * 32 + lanehi * 16;
            const v16h afr0 =
                load_afrag(&lds_outs[(mgrp * 32 + lanelo) * 256 + k0]);
            const v16h afr1 =
                load_afrag(&lds_outs[(mgrp * 32 + 16 + lanelo) * 256 + k0]);
            const int kg = half * 256 + ks * 32 + lanehi * 16;
            #pragma unroll
            for (int nt = 0; nt < 4; ++nt) {
                const int ncol = nq * 64 + nt * 16 + lanelo;
                const v16h bf = load16h(WLb + ncol * S_UNITS + kg);
                acc[nt]     = wmma16(afr0, bf, acc[nt]);
                acc[4 + nt] = wmma16(afr1, bf, acc[4 + nt]);
            }
        }
        __syncthreads();   // before next half overwrites lds_outs
    }

    // ---- write f32 output ----
    #pragma unroll
    for (int nt = 0; nt < 4; ++nt) {
        const int col = nq * 64 + nt * 16 + lanelo;
        #pragma unroll
        for (int mt = 0; mt < 2; ++mt) {
            #pragma unroll
            for (int r = 0; r < 8; ++r) {
                const int row = rowbase + mgrp * 32 + mt * 16 + r + lanehi * 8;
                out[row * OUT_BITS + col] = acc[mt * 4 + nt][r];
            }
        }
    }
}

// ---------------------------------------------------------------------------
extern "C" void kernel_launch(void* const* d_in, const int* in_sizes, int n_in,
                              void* d_out, int out_size, void* d_ws, size_t ws_size,
                              hipStream_t stream) {
    const float* x      = (const float*)d_in[0];
    const float* PL     = (const float*)d_in[1];
    const float* PR     = (const float*)d_in[2];
    const float* Wunits = (const float*)d_in[3];
    const float* WL     = (const float*)d_in[4];
    float* out = (float*)d_out;

    // Workspace carve-up (~793 KB needed).
    unsigned short* Wa  = (unsigned short*)d_ws;
    unsigned short* Wb  = Wa  + S_UNITS * B_IN;
    unsigned short* WLb = Wb  + S_UNITS * B_IN;
    float* C0 = (float*)(WLb + OUT_BITS * S_UNITS);
    float* C1 = C0 + S_UNITS;
    float* C2 = C1 + S_UNITS;

    prep_pairs<<<S_UNITS, 256, 0, stream>>>(PL, PR, Wunits, Wa, Wb, C0, C1, C2);
    prep_wl<<<OUT_BITS, 256, 0, stream>>>(WL, WLb);
    glayer_fused<<<BATCH / 64, 256, 0, stream>>>(x, Wa, Wb, WLb, C0, C1, C2, out);
}
